// TUPEMultiHeadAttention_13881334300859
// MI455X (gfx1250) — matrix operand order repaired and checked by reading, not compiled, not run on hardware
//
#include <hip/hip_runtime.h>
#include <hip/hip_bf16.h>
#include <math.h>
#include <stdint.h>

// ---------------------------------------------------------------------------
// TUPE multi-head attention for MI455X (gfx1250), wave32 + WMMA bf16 + TDM.
//   qh = q@Wq+bq ; kh = k@Wk+bk ; vh = v@Wv+bv
//   scores = (qh khT + gamma * qh pos_bias) / 8 -> online softmax -> @ vh
//   out = attn @ Wo + bo  (f32)
// GEMMs: v_wmma_f32_16x16x32_bf16; tile staging: tensor_load_to_lds (TDM),
// double buffered on TENSORcnt. s_wait_tensorcnt executed by all waves
// (no-op for non-issuing waves; avoids per-iteration scalar branch chains).
// ---------------------------------------------------------------------------

typedef __bf16 bf16_t;
typedef __attribute__((ext_vector_type(16))) __bf16 bf16x16;
typedef __attribute__((ext_vector_type(8)))  float  floatx8;
typedef __attribute__((ext_vector_type(4)))  unsigned int uint32x4;
typedef __attribute__((ext_vector_type(8)))  int int32x8;
typedef __attribute__((ext_vector_type(4)))  int int32x4;

#define BATCH     4
#define SEQ       1024
#define DMODEL    1024
#define NHEADS    16
#define DEPTH     64
#define MAXLEN    5000
#define POS_ROWS  (2 * MAXLEN - 1)   // 9999
#define NTOK      (BATCH * SEQ)      // 4096

__device__ __forceinline__ unsigned short bf16_bits(float f) {
  unsigned u = __float_as_uint(f);
  u = (u + 0x7FFFu + ((u >> 16) & 1u)) >> 16;   // round-to-nearest-even
  return (unsigned short)u;
}
__device__ __forceinline__ bf16_t to_bf16(float f) {
  unsigned short s = bf16_bits(f);
  bf16_t b;
  __builtin_memcpy(&b, &s, 2);
  return b;
}
__device__ __forceinline__ bf16x16 make_frag(uint4 lo, uint4 hi) {
  union { uint4 q[2]; bf16x16 v; } u;
  u.q[0] = lo; u.q[1] = hi;
  return u.v;
}
__device__ __forceinline__ floatx8 wmma_bf16(bf16x16 a, bf16x16 b, floatx8 c) {
  return __builtin_amdgcn_wmma_f32_16x16x32_bf16(false, a, false, b,
                                                 (short)0, c, false, false);
}

// ---------------------------------------------------------------------------
// TDM 2-D tile load: global (bf16, row stride = stride_elems) -> LDS with
// row padding. D# packing per CDNA5 ISA.
//  pad_int_code: DWORDs-per-row code (2<<code DW);  pad_amt_code: pad DW - 1.
// ---------------------------------------------------------------------------
__device__ __forceinline__ void tdm_load_2d(unsigned lds_addr, const void* gptr,
                                            unsigned tile_x, unsigned tile_y,
                                            unsigned stride_elems,
                                            unsigned pad_int_code,
                                            unsigned pad_amt_code,
                                            unsigned tdim0, unsigned tdim1) {
  uint64_t ga = (uint64_t)(uintptr_t)gptr;
  uint32x4 g0;
  g0[0] = 1u;                                            // count=1, user mode
  g0[1] = lds_addr;                                      // LDS byte address
  g0[2] = (unsigned)ga;                                  // global addr [31:0]
  g0[3] = (unsigned)((ga >> 32) & 0x01FFFFFFu) | 0x80000000u; // [56:32]|type=2
  int32x8 g1;
  g1[0] = (int)((1u << 16) |                             // data_size = 2B
                (1u << 20) |                             // pad_enable
                (pad_int_code << 22) | (pad_amt_code << 25));
  g1[1] = (int)((tdim0 & 0xFFFFu) << 16);                // tensor_dim0 lo16
  g1[2] = (int)(((tdim0 >> 16) & 0xFFFFu) |              // tensor_dim0 hi16
                ((tdim1 & 0xFFFFu) << 16));              // tensor_dim1 lo16
  g1[3] = (int)(((tdim1 >> 16) & 0xFFFFu) |              // tensor_dim1 hi16
                ((tile_x & 0xFFFFu) << 16));             // tile_dim0
  g1[4] = (int)(tile_y & 0xFFFFu);                       // tile_dim1 (dim2=0)
  g1[5] = (int)stride_elems;                             // dim0_stride lo32
  g1[6] = 0;
  g1[7] = 0;
  int32x4 z4 = {0, 0, 0, 0};
  int32x8 z8 = {0, 0, 0, 0, 0, 0, 0, 0};
  __builtin_amdgcn_tensor_load_to_lds(g0, g1, z4, z4, z8, 0);
}

__device__ __forceinline__ unsigned lds_off32(const void* p) {
  return (unsigned)(uintptr_t)p;   // low 32 bits of generic ptr = LDS offset
}

// ---------------------------------------------------------------------------
// f32 -> bf16 elementwise convert
// ---------------------------------------------------------------------------
__global__ void k_convert(const float* __restrict__ src,
                          bf16_t* __restrict__ dst, int n) {
  int i = blockIdx.x * blockDim.x + threadIdx.x;
  if (i < n) dst[i] = to_bf16(src[i]);
}

// ---------------------------------------------------------------------------
// f32 [R,C] -> bf16 transposed [C,R]   (32x32 LDS tile, block 256)
// ---------------------------------------------------------------------------
__global__ void __launch_bounds__(256)
k_convert_t(const float* __restrict__ src, bf16_t* __restrict__ dst,
            int R, int C) {
  __shared__ float tile[32][33];
  const int r0 = blockIdx.y * 32, c0 = blockIdx.x * 32;
  const int tx = threadIdx.x & 31, ty = threadIdx.x >> 5;
#pragma unroll
  for (int yy = 0; yy < 4; ++yy)
    tile[ty + yy * 8][tx] = src[(size_t)(r0 + ty + yy * 8) * C + c0 + tx];
  __syncthreads();
#pragma unroll
  for (int yy = 0; yy < 4; ++yy)
    dst[(size_t)(c0 + ty + yy * 8) * R + r0 + tx] = to_bf16(tile[tx][ty + yy * 8]);
}

// ---------------------------------------------------------------------------
// bf16 GEMM: C[M,N] = A[M,K] @ W[K,N] + bias[N], W given pre-transposed
// Wt[N,K]. Block 256 thr (8 waves), tile 128x128, BK=32; TDM double buffer.
//   mode 0: bf16 head-split  out[((b*H+h)*S+s)*64+dd]
//   mode 1: f32 row-major    out[m*N+n]
//   mode 2: bf16 V-transposed out[((b*H+h)*64+dd)*S+s]
// ---------------------------------------------------------------------------
__global__ void __launch_bounds__(256)
k_gemm(const bf16_t* __restrict__ A, const bf16_t* __restrict__ Wt,
       const float* __restrict__ bias, void* __restrict__ outp,
       int M, int N, int K, int mode) {
  __shared__ __align__(16) unsigned short Asub[2][128 * 40]; // [row][k] pad 40
  __shared__ __align__(16) unsigned short Bt[2][128 * 40];   // [col][k] pad 40

  const int tid = threadIdx.x;
  const int l   = tid & 31;
  const int ln  = l & 15;
  const int hi  = l >> 4;
  const int w   = tid >> 5;
  const int wm  = w >> 1;          // 0..3
  const int wn  = w & 1;           // 0..1
  const int m0  = blockIdx.y * 128;
  const int n0  = blockIdx.x * 128;
  const unsigned wid = (unsigned)__builtin_amdgcn_readfirstlane(tid >> 5);

  floatx8 acc[2][4];
  const floatx8 fz = {0.f, 0.f, 0.f, 0.f, 0.f, 0.f, 0.f, 0.f};
#pragma unroll
  for (int ms = 0; ms < 2; ++ms)
#pragma unroll
    for (int ns = 0; ns < 4; ++ns) acc[ms][ns] = fz;

  const int nk = K / 32;
  // prologue: stage tile 0 (A row = 32 bf16 = 16 DW -> pad code 3; +4 DW pad)
  if (wid == 0) {
    tdm_load_2d(lds_off32(&Asub[0][0]), A + (size_t)m0 * K, 32, 128,
                (unsigned)K, 3, 3, (unsigned)K, (unsigned)M);
    tdm_load_2d(lds_off32(&Bt[0][0]), Wt + (size_t)n0 * K, 32, 128,
                (unsigned)K, 3, 3, (unsigned)K, (unsigned)N);
  }

  for (int it = 0; it < nk; ++it) {
    const int cur = it & 1;
    if (it + 1 < nk) {
      if (wid == 0) {
        const int k1 = (it + 1) * 32;
        tdm_load_2d(lds_off32(&Asub[cur ^ 1][0]), A + (size_t)m0 * K + k1,
                    32, 128, (unsigned)K, 3, 3, (unsigned)K, (unsigned)M);
        tdm_load_2d(lds_off32(&Bt[cur ^ 1][0]), Wt + (size_t)n0 * K + k1,
                    32, 128, (unsigned)K, 3, 3, (unsigned)K, (unsigned)N);
      }
      __builtin_amdgcn_s_wait_tensorcnt(2);   // tile `it` complete (wave 0);
    } else {                                  // no-op for other waves
      __builtin_amdgcn_s_wait_tensorcnt(0);
    }
    __syncthreads();

    bf16x16 af[2], bfr[4];
#pragma unroll
    for (int ms = 0; ms < 2; ++ms) {
      int row = wm * 32 + ms * 16 + ln;
      const unsigned short* p = &Asub[cur][row * 40 + hi * 16];
      af[ms] = make_frag(*(const uint4*)p, *(const uint4*)(p + 8));
    }
#pragma unroll
    for (int ns = 0; ns < 4; ++ns) {
      int col = wn * 64 + ns * 16 + ln;
      const unsigned short* p = &Bt[cur][col * 40 + hi * 16];
      bfr[ns] = make_frag(*(const uint4*)p, *(const uint4*)(p + 8));
    }
#pragma unroll
    for (int ms = 0; ms < 2; ++ms)
#pragma unroll
      for (int ns = 0; ns < 4; ++ns)
        acc[ms][ns] = wmma_bf16(af[ms], bfr[ns], acc[ms][ns]);
    __syncthreads();   // all reads done before buffer reuse
  }

  // ---- epilogue ----
#pragma unroll
  for (int ms = 0; ms < 2; ++ms) {
#pragma unroll
    for (int ns = 0; ns < 4; ++ns) {
      int n = n0 + wn * 64 + ns * 16 + ln;
      float bv = bias[n];
#pragma unroll
      for (int r = 0; r < 8; ++r) {
        int m = m0 + wm * 32 + ms * 16 + r + hi * 8;
        float v = acc[ms][ns][r] + bv;
        int b = m >> 10, s = m & (SEQ - 1);
        int h = n >> 6, dd = n & 63;
        if (mode == 0) {
          ((bf16_t*)outp)[((size_t)(b * NHEADS + h) * SEQ + s) * DEPTH + dd] =
              to_bf16(v);
        } else if (mode == 1) {
          ((float*)outp)[(size_t)m * N + n] = v;
        } else {
          ((bf16_t*)outp)[((size_t)(b * NHEADS + h) * DEPTH + dd) * SEQ + s] =
              to_bf16(v);
        }
      }
    }
  }
}

// ---------------------------------------------------------------------------
// Flash attention with relative positional bias.
// Grid: (B*H, S/64). Block: 128 thr = 4 waves, each wave owns 16 query rows.
// V arrives pre-transposed Vtg[B,H,64,S]; per 64-key tile it is TDM-copied
// into LDS while the G-band (rel-bias) and QK^T WMMAs run.
// ---------------------------------------------------------------------------
__global__ void __launch_bounds__(128)
k_attn(const bf16_t* __restrict__ Qh, const bf16_t* __restrict__ Kh,
       const bf16_t* __restrict__ Vtg, const bf16_t* __restrict__ posb,
       const float* __restrict__ gamma, bf16_t* __restrict__ attn_out) {
  __shared__ __align__(16) unsigned short Vt[64 * 72];    // [dd][key] pad 72
  __shared__ __align__(16) float Gbuf[4][16 * 80];        // per-wave G band
  __shared__ __align__(16) unsigned short Pl[4][16 * 72]; // per-wave P

  const int tid = threadIdx.x;
  const int l   = tid & 31, w = tid >> 5;
  const int ln  = l & 15,  hi = l >> 4;
  const int bh  = blockIdx.x;             // b*16 + h
  const int q0w = blockIdx.y * 64 + w * 16;
  const int b   = bh >> 4, h = bh & 15;
  const float g = gamma[h];
  const size_t headbase = (size_t)bh * SEQ * DEPTH;
  const unsigned wid = (unsigned)__builtin_amdgcn_readfirstlane(tid >> 5);

  // Q fragments (resident): two K=32 steps over DEPTH=64
  bf16x16 qa[2];
  {
    const bf16_t* qrow = Qh + headbase + (size_t)(q0w + ln) * DEPTH;
#pragma unroll
    for (int t = 0; t < 2; ++t) {
      const bf16_t* p = qrow + t * 32 + hi * 16;
      qa[t] = make_frag(*(const uint4*)p, *(const uint4*)(p + 8));
    }
  }

  const floatx8 fz = {0.f, 0.f, 0.f, 0.f, 0.f, 0.f, 0.f, 0.f};
  floatx8 O[4];
#pragma unroll
  for (int nd = 0; nd < 4; ++nd) O[nd] = fz;
  float m_i[8], l_i[8];
#pragma unroll
  for (int r = 0; r < 8; ++r) { m_i[r] = -1e30f; l_i[r] = 0.f; }

  for (int kt0 = 0; kt0 < SEQ; kt0 += 64) {
    __syncthreads();   // prior V-tile readers done
    // ---- TDM: stage V tile Vt[dd][kt0..kt0+63]  (row 64 bf16 = 32 DW) ----
    if (wid == 0)
      tdm_load_2d(lds_off32(&Vt[0]),
                  Vtg + (size_t)bh * DEPTH * SEQ + kt0,
                  64, 64, SEQ, 4, 3, SEQ, DEPTH);

    // ---- relative-bias band GEMM: G[i,t] -> Gbuf (overlaps TDM) ----
    const int prbase = q0w - kt0 + (MAXLEN - 1) - 63;
#pragma unroll
    for (int gt = 0; gt < 5; ++gt) {
      floatx8 gacc = fz;
      const bf16_t* pp = posb + (size_t)(prbase + gt * 16 + ln) * DEPTH;
#pragma unroll
      for (int t = 0; t < 2; ++t) {
        const bf16_t* p = pp + t * 32 + hi * 16;
        bf16x16 pf = make_frag(*(const uint4*)p, *(const uint4*)(p + 8));
        gacc = wmma_bf16(qa[t], pf, gacc);
      }
#pragma unroll
      for (int r = 0; r < 8; ++r)
        Gbuf[w][(r + hi * 8) * 80 + gt * 16 + ln] = gacc[r];
    }
    // ---- content scores: Q . K^T (overlaps TDM) ----
    floatx8 sacc[4];
#pragma unroll
    for (int ns = 0; ns < 4; ++ns) {
      sacc[ns] = fz;
      const bf16_t* krow = Kh + headbase + (size_t)(kt0 + ns * 16 + ln) * DEPTH;
#pragma unroll
      for (int t = 0; t < 2; ++t) {
        const bf16_t* p = krow + t * 32 + hi * 16;
        bf16x16 kf = make_frag(*(const uint4*)p, *(const uint4*)(p + 8));
        sacc[ns] = wmma_bf16(qa[t], kf, sacc[ns]);
      }
    }
    __builtin_amdgcn_s_wait_tensorcnt(0);   // wave0: Vt done; others: no-op
    __syncthreads();                        // Vt staged, Gbuf visible

    // ---- add rel bias, scale; online softmax over this tile ----
#pragma unroll
    for (int ns = 0; ns < 4; ++ns) {
      int j = ns * 16 + ln;
#pragma unroll
      for (int r = 0; r < 8; ++r) {
        int i = r + hi * 8;
        float cps = Gbuf[w][i * 80 + (i - j + 63)];
        sacc[ns][r] = (sacc[ns][r] + g * cps) * 0.125f;
      }
    }
    float osc[8];
#pragma unroll
    for (int r = 0; r < 8; ++r) {
      float rm = fmaxf(fmaxf(sacc[0][r], sacc[1][r]),
                       fmaxf(sacc[2][r], sacc[3][r]));
#pragma unroll
      for (int off = 1; off < 16; off <<= 1)
        rm = fmaxf(rm, __shfl_xor(rm, off, 16));
      float mn = fmaxf(m_i[r], rm);
      float sc = __expf(m_i[r] - mn);
      m_i[r] = mn;
      float rs = 0.f;
#pragma unroll
      for (int ns = 0; ns < 4; ++ns) {
        float p = __expf(sacc[ns][r] - mn);
        sacc[ns][r] = p;
        rs += p;
      }
#pragma unroll
      for (int off = 1; off < 16; off <<= 1)
        rs += __shfl_xor(rs, off, 16);
      l_i[r] = l_i[r] * sc + rs;
      osc[r] = sc;
    }
#pragma unroll
    for (int nd = 0; nd < 4; ++nd)
#pragma unroll
      for (int r = 0; r < 8; ++r) O[nd][r] *= osc[r];

    // ---- P -> LDS (C-layout to A-layout round trip) ----
#pragma unroll
    for (int ns = 0; ns < 4; ++ns)
#pragma unroll
      for (int r = 0; r < 8; ++r)
        Pl[w][(r + hi * 8) * 72 + ns * 16 + ln] = bf16_bits(sacc[ns][r]);
    __syncthreads();

    // ---- O += P @ V ----
    bf16x16 pf2[2];
#pragma unroll
    for (int t = 0; t < 2; ++t) {
      const unsigned short* p = &Pl[w][ln * 72 + t * 32 + hi * 16];
      pf2[t] = make_frag(*(const uint4*)p, *(const uint4*)(p + 8));
    }
#pragma unroll
    for (int nd = 0; nd < 4; ++nd) {
#pragma unroll
      for (int t = 0; t < 2; ++t) {
        const unsigned short* p = &Vt[(nd * 16 + ln) * 72 + t * 32 + hi * 16];
        bf16x16 vf = make_frag(*(const uint4*)p, *(const uint4*)(p + 8));
        O[nd] = wmma_bf16(pf2[t], vf, O[nd]);
      }
    }
  }

  // ---- normalize + store merged-head bf16 [B,S,D] ----
#pragma unroll
  for (int nd = 0; nd < 4; ++nd) {
#pragma unroll
    for (int r = 0; r < 8; ++r) {
      int i = r + hi * 8;
      float v = O[nd][r] / l_i[r];
      attn_out[(size_t)(b * SEQ + q0w + i) * DMODEL + h * 64 + nd * 16 + ln] =
          to_bf16(v);
    }
  }
}

// ---------------------------------------------------------------------------
extern "C" void kernel_launch(void* const* d_in, const int* in_sizes, int n_in,
                              void* d_out, int out_size, void* d_ws,
                              size_t ws_size, hipStream_t stream) {
  const float* v_f   = (const float*)d_in[0];
  const float* k_f   = (const float*)d_in[1];
  const float* q_f   = (const float*)d_in[2];
  const float* Wq    = (const float*)d_in[3];
  const float* bq    = (const float*)d_in[4];
  const float* Wk    = (const float*)d_in[5];
  const float* bk    = (const float*)d_in[6];
  const float* Wv    = (const float*)d_in[7];
  const float* bv    = (const float*)d_in[8];
  const float* pos   = (const float*)d_in[9];
  const float* gamma = (const float*)d_in[10];
  const float* Wo    = (const float*)d_in[11];
  const float* bo    = (const float*)d_in[12];

  char* ws = (char*)d_ws;
  size_t off = 0;
  auto alloc = [&](size_t bytes) -> void* {
    void* p = ws + off;
    off += (bytes + 255) & ~(size_t)255;
    return p;
  };

  const int NELEM = NTOK * DMODEL;           // 4M
  const int WELEM = DMODEL * DMODEL;         // 1M
  const int PELEM = POS_ROWS * DEPTH;        // ~640k

  bf16_t* qb   = (bf16_t*)alloc((size_t)NELEM * 2);
  bf16_t* kb   = (bf16_t*)alloc((size_t)NELEM * 2);
  bf16_t* vb   = (bf16_t*)alloc((size_t)NELEM * 2);
  bf16_t* Wqt  = (bf16_t*)alloc((size_t)WELEM * 2);   // transposed [N,K]
  bf16_t* Wkt  = (bf16_t*)alloc((size_t)WELEM * 2);
  bf16_t* Wvt  = (bf16_t*)alloc((size_t)WELEM * 2);
  bf16_t* Wot  = (bf16_t*)alloc((size_t)WELEM * 2);
  bf16_t* posb = (bf16_t*)alloc((size_t)PELEM * 2);
  bf16_t* Qhb  = (bf16_t*)alloc((size_t)NELEM * 2);   // [B,H,S,64]
  bf16_t* Khb  = (bf16_t*)alloc((size_t)NELEM * 2);   // [B,H,S,64]
  bf16_t* Vtg  = (bf16_t*)alloc((size_t)NELEM * 2);   // [B,H,64,S]
  bf16_t* atnb = (bf16_t*)alloc((size_t)NELEM * 2);   // [B,S,D]

  // converts
  k_convert<<<(NELEM + 255) / 256, 256, 0, stream>>>(q_f, qb, NELEM);
  k_convert<<<(NELEM + 255) / 256, 256, 0, stream>>>(k_f, kb, NELEM);
  k_convert<<<(NELEM + 255) / 256, 256, 0, stream>>>(v_f, vb, NELEM);
  k_convert<<<(PELEM + 255) / 256, 256, 0, stream>>>(pos, posb, PELEM);
  dim3 gt(DMODEL / 32, DMODEL / 32);
  k_convert_t<<<gt, 256, 0, stream>>>(Wq, Wqt, DMODEL, DMODEL);
  k_convert_t<<<gt, 256, 0, stream>>>(Wk, Wkt, DMODEL, DMODEL);
  k_convert_t<<<gt, 256, 0, stream>>>(Wv, Wvt, DMODEL, DMODEL);
  k_convert_t<<<gt, 256, 0, stream>>>(Wo, Wot, DMODEL, DMODEL);

  // projections
  dim3 gg(DMODEL / 128, NTOK / 128);   // (8, 32)
  k_gemm<<<gg, 256, 0, stream>>>(qb, Wqt, bq, Qhb, NTOK, DMODEL, DMODEL, 0);
  k_gemm<<<gg, 256, 0, stream>>>(kb, Wkt, bk, Khb, NTOK, DMODEL, DMODEL, 0);
  k_gemm<<<gg, 256, 0, stream>>>(vb, Wvt, bv, Vtg, NTOK, DMODEL, DMODEL, 2);

  // flash attention with relative positional bias
  dim3 ga(BATCH * NHEADS, SEQ / 64);   // (64, 16)
  k_attn<<<ga, 128, 0, stream>>>(Qhb, Khb, Vtg, posb, gamma, atnb);

  // output projection -> f32 d_out
  k_gemm<<<gg, 256, 0, stream>>>(atnb, Wot, bo, d_out, NTOK, DMODEL, DMODEL, 1);
}
